// MemoryUnit_3693671874649
// MI455X (gfx1250) — matrix operand order repaired and checked by reading, not compiled, not run on hardware
//
#include <hip/hip_runtime.h>

#define N_ROWS   65536
#define M_DIM    2000
#define F_DIM    512
#define M_PAD    2016          // 63 * 32, zero-padded K for GEMM2
#define ATT_STRIDE 2048        // padded LDS row stride (elements)
#define LAMBDA   0.0025f
#define EPS_     1e-12f

typedef __attribute__((ext_vector_type(16))) __bf16 v16bf;
typedef __attribute__((ext_vector_type(8)))  float  v8f;

__device__ __forceinline__ unsigned short f2bf(float x) {
    unsigned int u = __builtin_bit_cast(unsigned int, x);
    return (unsigned short)((u + 0x7FFFu + ((u >> 16) & 1u)) >> 16); // RNE
}

union Frag16 { unsigned int u[8]; uint4 q[2]; v16bf v; };

// ---------------------------------------------------------------------------
// Kernel 0: pack weight into bf16, row-major Wbf[2000][512] and transposed,
// K-padded WbfT[512][2016] (zeros for m >= 2000).
// ---------------------------------------------------------------------------
__global__ void pack_weight(const float* __restrict__ w,
                            unsigned short* __restrict__ Wbf,
                            unsigned short* __restrict__ WbfT) {
    int idx = blockIdx.x * blockDim.x + threadIdx.x;
    if (idx >= F_DIM * M_PAD) return;
    int f = idx / M_PAD;
    int m = idx - f * M_PAD;
    float val = (m < M_DIM) ? w[(size_t)m * F_DIM + f] : 0.0f;
    unsigned short b = f2bf(val);
    WbfT[(size_t)f * M_PAD + m] = b;
    if (m < M_DIM) Wbf[(size_t)m * F_DIM + f] = b;
}

// ---------------------------------------------------------------------------
// Kernel 1: fused  att = softmax(x @ W^T) -> hardshrink -> L1 norm
//                  out = att @ W
// One block = 16 rows of x; att row-stripe lives entirely in LDS.
// LDS: xbf[16][512] bf16 | att[16][2048] f32 | attbf[16][2048] bf16
// ---------------------------------------------------------------------------
__global__ void __launch_bounds__(256)
fused_memory_unit(const float* __restrict__ x,
                  const unsigned short* __restrict__ Wbf,   // [2000][512] bf16
                  const unsigned short* __restrict__ WbfT,  // [512][2016] bf16
                  float* __restrict__ out,                  // [N][512]
                  float* __restrict__ attOut)               // [N][2000]
{
    extern __shared__ char smem[];
    unsigned short* xbf   = (unsigned short*)smem;                         // 16 KB
    float*          att   = (float*)(smem + 16 * F_DIM * 2);               // 128 KB
    unsigned short* attbf = (unsigned short*)(smem + 16 * F_DIM * 2
                                                   + 16 * ATT_STRIDE * 4); // 64 KB

    const int tid  = threadIdx.x;
    const int lane = tid & 31;
    const int wave = tid >> 5;          // 8 waves
    const int n0   = blockIdx.x * 16;

    const int half = lane >> 4;         // 0: lanes 0-15, 1: lanes 16-31
    const int mrow = lane & 15;

    // ---- stage 16 x-rows into LDS as bf16 ----
    for (int i = tid; i < 16 * F_DIM; i += 256) {
        int r = i >> 9, c = i & 511;
        xbf[i] = f2bf(x[(size_t)(n0 + r) * F_DIM + c]);
    }
    __syncthreads();

    // ---- GEMM1: att[16 x 2000] = xbf @ W^T (bf16 WMMA, f32 acc) ----
    for (int mt = wave; mt < M_DIM / 16; mt += 8) {
        v8f acc = {0.f,0.f,0.f,0.f,0.f,0.f,0.f,0.f};
        const int wr = mt * 16 + mrow;              // weight row = this lane's B column
        for (int kt = 0; kt < F_DIM / 32; ++kt) {
            const int f0 = kt * 32;
            // A fragment: lane holds 16 consecutive k starting at half * 16 (matches B)
            Frag16 a;
            #pragma unroll
            for (int p = 0; p < 8; ++p) {
                int k0 = f0 + half * 16 + 2 * p;
                a.u[p] = *(const unsigned int*)&xbf[mrow * F_DIM + k0];
            }
            // B fragment: lane holds 16 consecutive f of weight row wr
            Frag16 b;
            const uint4* bp = (const uint4*)(Wbf + (size_t)wr * F_DIM + f0 + half * 16);
            b.q[0] = bp[0];
            b.q[1] = bp[1];
            __builtin_prefetch(Wbf + (size_t)wr * F_DIM + f0 + 64, 0, 3);
            acc = __builtin_amdgcn_wmma_f32_16x16x32_bf16(false, a.v, false, b.v,
                                                          (short)0, acc, false, false);
        }
        #pragma unroll
        for (int r = 0; r < 8; ++r)
            att[(r + 8 * half) * ATT_STRIDE + mt * 16 + mrow] = acc[r];
    }
    __syncthreads();

    // ---- softmax + hardshrink-relu + L1 normalize ----
    // one row per 16-lane group, pair-wise (each lane owns m = 2p, 2p+1)
    {
        const int g = tid >> 4;                     // row 0..15
        const int l = tid & 15;
        float* arow = att + g * ATT_STRIDE;
        unsigned int* brow = (unsigned int*)(attbf + g * ATT_STRIDE);

        float mx = -3.4e38f;
        for (int p = l; p < M_DIM / 2; p += 16) {
            float2 v = *(const float2*)&arow[2 * p];
            mx = fmaxf(mx, fmaxf(v.x, v.y));
        }
        #pragma unroll
        for (int o = 8; o >= 1; o >>= 1) mx = fmaxf(mx, __shfl_xor(mx, o, 16));

        float s = 0.f;
        for (int p = l; p < M_DIM / 2; p += 16) {
            float2 v = *(const float2*)&arow[2 * p];
            v.x = __expf(v.x - mx);
            v.y = __expf(v.y - mx);
            *(float2*)&arow[2 * p] = v;
            s += v.x + v.y;
        }
        #pragma unroll
        for (int o = 8; o >= 1; o >>= 1) s += __shfl_xor(s, o, 16);
        const float inv = 1.f / s;

        float l1 = 0.f;
        for (int p = l; p < M_DIM / 2; p += 16) {
            float2 v = *(const float2*)&arow[2 * p];
            float v0 = v.x * inv, v1 = v.y * inv;
            float t0 = v0 - LAMBDA, t1 = v1 - LAMBDA;
            float s0 = fmaxf(t0, 0.f) * v0 / (fabsf(t0) + EPS_);
            float s1 = fmaxf(t1, 0.f) * v1 / (fabsf(t1) + EPS_);
            *(float2*)&arow[2 * p] = make_float2(s0, s1);
            l1 += s0 + s1;
        }
        #pragma unroll
        for (int o = 8; o >= 1; o >>= 1) l1 += __shfl_xor(l1, o, 16);
        const float invl1 = 1.f / fmaxf(l1, EPS_);

        for (int p = l; p < M_DIM / 2; p += 16) {
            float2 v = *(const float2*)&arow[2 * p];
            v.x *= invl1;
            v.y *= invl1;
            // second output (f32), 8-byte aligned pair store
            *(float2*)&attOut[(size_t)(n0 + g) * M_DIM + 2 * p] = v;
            // packed bf16 A-operand for GEMM2
            brow[p] = (unsigned int)f2bf(v.x) | ((unsigned int)f2bf(v.y) << 16);
        }
        if (l < 8) brow[M_DIM / 2 + l] = 0u;        // zero K-pad m in [2000, 2016)
    }
    __syncthreads();

    // ---- GEMM2: out[16 x 512] = att @ W ----
    // 4 accumulators per wave; A fragment built once per kt, reused 4x.
    {
        const int ftb = wave * 4;                   // 8 waves x 4 = 32 f-tiles
        v8f acc[4];
        #pragma unroll
        for (int j = 0; j < 4; ++j) acc[j] = (v8f){0.f,0.f,0.f,0.f,0.f,0.f,0.f,0.f};

        for (int kt = 0; kt < M_PAD / 32; ++kt) {
            const int m0 = kt * 32;
            Frag16 a;
            #pragma unroll
            for (int p = 0; p < 8; ++p) {
                int k0 = m0 + half * 16 + 2 * p;
                a.u[p] = *(const unsigned int*)&attbf[mrow * ATT_STRIDE + k0];
            }
            #pragma unroll
            for (int j = 0; j < 4; ++j) {
                const int fcol = (ftb + j) * 16 + mrow;
                Frag16 b;
                const uint4* bp = (const uint4*)(WbfT + (size_t)fcol * M_PAD + m0 + half * 16);
                b.q[0] = bp[0];
                b.q[1] = bp[1];
                acc[j] = __builtin_amdgcn_wmma_f32_16x16x32_bf16(false, a.v, false, b.v,
                                                                 (short)0, acc[j], false, false);
            }
        }
        #pragma unroll
        for (int j = 0; j < 4; ++j) {
            const int fcol = (ftb + j) * 16 + mrow;
            #pragma unroll
            for (int r = 0; r < 8; ++r)
                out[(size_t)(n0 + r + 8 * half) * F_DIM + fcol] = acc[j][r];
        }
    }
}

// ---------------------------------------------------------------------------
extern "C" void kernel_launch(void* const* d_in, const int* in_sizes, int n_in,
                              void* d_out, int out_size, void* d_ws, size_t ws_size,
                              hipStream_t stream) {
    const float* x = (const float*)d_in[0];   // [65536, 512]
    const float* w = (const float*)d_in[1];   // [2000, 512]

    float* out    = (float*)d_out;                          // [N, 512]
    float* attOut = out + (size_t)N_ROWS * F_DIM;           // [N, 2000]

    unsigned short* Wbf  = (unsigned short*)d_ws;                   // 2000*512 bf16
    unsigned short* WbfT = Wbf + (size_t)M_DIM * F_DIM;             // 512*2016 bf16

    {
        int total = F_DIM * M_PAD;
        pack_weight<<<(total + 255) / 256, 256, 0, stream>>>(w, Wbf, WbfT);
    }

    size_t smem = (size_t)16 * F_DIM * sizeof(unsigned short)       // xbf    16 KB
                + (size_t)16 * ATT_STRIDE * sizeof(float)           // att   128 KB
                + (size_t)16 * ATT_STRIDE * sizeof(unsigned short); // attbf  64 KB
    fused_memory_unit<<<N_ROWS / 16, 256, smem, stream>>>(x, Wbf, WbfT, out, attOut);
}
